// SimpleTransformerBlock_38835094290561
// MI455X (gfx1250) — matrix operand order repaired and checked by reading, not compiled, not run on hardware
//
#include <hip/hip_runtime.h>

// ---------------- problem constants ----------------
#define DIM   1024
#define HDIM  4096
#define NH    16
#define HD    64
#define SEQ   2048
#define BATCH 2
#define TOK   (BATCH*SEQ)   // 4096 tokens

// ---------------- WMMA types ----------------
typedef __attribute__((ext_vector_type(16))) __bf16 v16bf;
typedef __attribute__((ext_vector_type(8)))  float  v8f;
typedef __attribute__((ext_vector_type(4)))  int    v4i;

union Frag { v16bf v; unsigned u[8]; };

// ---------------- optional CDNA5 async-to-LDS path ----------------
#if __has_builtin(__builtin_amdgcn_global_load_async_to_lds_b128) && \
    __has_builtin(__builtin_amdgcn_s_wait_asynccnt)
#define HAVE_ASYNC 1
#else
#define HAVE_ASYNC 0
#endif

__device__ __forceinline__ void async_copy_b128(void* dst_lds, const void* src_global) {
#if HAVE_ASYNC
    __builtin_amdgcn_global_load_async_to_lds_b128(
        (__attribute__((address_space(1))) v4i*)src_global,
        (__attribute__((address_space(3))) v4i*)dst_lds,
        0, 0);
#else
    *(uint4*)dst_lds = *(const uint4*)src_global;
#endif
}

__device__ __forceinline__ void wait_async_copies() {
#if HAVE_ASYNC
    __builtin_amdgcn_s_wait_asynccnt(0);
#endif
}

__device__ __forceinline__ v8f vzero8() {
    v8f z;
#pragma unroll
    for (int i = 0; i < 8; ++i) z[i] = 0.0f;
    return z;
}

__device__ __forceinline__ unsigned short bf16rne(float f) {
    unsigned u = __float_as_uint(f);
    u += 0x7fffu + ((u >> 16) & 1u);       // round-to-nearest-even
    return (unsigned short)(u >> 16);
}

// A-fragment (16x32 bf16, MxK): lane m = lane&15, half = lane>>4 selects K phase.
__device__ __forceinline__ Frag load_afrag(const unsigned* base32, int idxBase, int half) {
    Frag a;
#pragma unroll
    for (int j = 0; j < 4; ++j) {
        a.u[j]     = base32[idxBase + half * 4 + j];
        a.u[4 + j] = base32[idxBase + 8 + half * 4 + j];
    }
    return a;
}

// B-fragment (32x16 bf16, KxN): lane n = lane&15; LDS holds B transposed [n][k] so
// the (2v,2v+1) K-pairs are contiguous 32-bit reads.
__device__ __forceinline__ Frag load_bfrag(const unsigned* base32, int idxBase, int half) {
    Frag b;
#pragma unroll
    for (int v = 0; v < 8; ++v) b.u[v] = base32[idxBase + (v >> 2) * 8 + half * 4 + (v & 3)];
    return b;
}

// ---------------- vectorized cast kernel (8 f32 -> 8 bf16 per thread) ----------------
__global__ void cast_f32_bf16_v8(const float4* __restrict__ in,
                                 uint4* __restrict__ out, int n8) {
    int i = blockIdx.x * blockDim.x + threadIdx.x;
    if (i >= n8) return;
    float4 a = in[i * 2 + 0];
    float4 b = in[i * 2 + 1];
    uint4 r;
    r.x = (unsigned)bf16rne(a.x) | ((unsigned)bf16rne(a.y) << 16);
    r.y = (unsigned)bf16rne(a.z) | ((unsigned)bf16rne(a.w) << 16);
    r.z = (unsigned)bf16rne(b.x) | ((unsigned)bf16rne(b.y) << 16);
    r.w = (unsigned)bf16rne(b.z) | ((unsigned)bf16rne(b.w) << 16);
    out[i] = r;
}

// ---------------- bf16 WMMA GEMM, BK=64, ping-pong double buffer ----------------
#define BM 128
#define BN 128
#define BK 64

__global__ __launch_bounds__(256)
void gemm_bf16_wmma(const unsigned short* __restrict__ A,
                    const unsigned short* __restrict__ B,
                    const float* __restrict__ residual,   // nullable [M,N] f32
                    float* __restrict__ outF,             // nullable [M,N] f32
                    unsigned short* __restrict__ outB,    // nullable bf16
                    int M, int N, int K,
                    int relu, int permute_qkv)
{
    __shared__ __attribute__((aligned(16))) unsigned short lds_a[2][BM * BK]; // [m][k]
    __shared__ __attribute__((aligned(16))) unsigned short lds_b[2][BN * BK]; // [n][k]

    const int tid  = threadIdx.x;
    const int wave = tid >> 5;
    const int lane = tid & 31;
    const int lm   = lane & 15;
    const int half = lane >> 4;
    const int wm   = wave & 3;   // M offset wm*32
    const int wn   = wave >> 2;  // N offset wn*64
    const int bm0  = blockIdx.y * BM;
    const int bn0  = blockIdx.x * BN;

    v8f acc[2][4];
#pragma unroll
    for (int i = 0; i < 2; ++i)
#pragma unroll
        for (int j = 0; j < 4; ++j) acc[i][j] = vzero8();

    auto stage = [&](int p, int kk) {
        // ---- A tile 128x64, identity layout: async b128 copies (1024 slots) ----
        uint4* sa = (uint4*)lds_a[p];
#pragma unroll
        for (int e = 0; e < 4; ++e) {
            int s   = e * 256 + tid;     // 8 uint4 per 64-elem row
            int row = s >> 3, ch = s & 7;
            async_copy_b128(&sa[s],
                            A + ((size_t)(bm0 + row) * K + kk + ch * 8));
        }
        // ---- B tile 64x128 -> [n][k] with interleaved K-pair b32 stores ----
        unsigned* sb = (unsigned*)lds_b[p];
#pragma unroll
        for (int e = 0; e < 2; ++e) {
            int tsk = e * 256 + tid;     // 32 k-pairs x 16 n-groups
            int kp  = tsk >> 4;
            int n0  = (tsk & 15) * 8;
            union { uint4 q; unsigned short s[8]; } r0, r1;
            r0.q = *(const uint4*)(B + (size_t)(kk + 2 * kp)     * N + bn0 + n0);
            r1.q = *(const uint4*)(B + (size_t)(kk + 2 * kp + 1) * N + bn0 + n0);
#pragma unroll
            for (int i = 0; i < 8; ++i)
                sb[(n0 + i) * (BK / 2) + kp] =
                    (unsigned)r0.s[i] | ((unsigned)r1.s[i] << 16);
        }
    };

    stage(0, 0);
    wait_async_copies();
    __syncthreads();

    int pb = 0;
    for (int kk = 0; kk < K; kk += BK, pb ^= 1) {
        if (kk + BK < K) stage(pb ^ 1, kk + BK);   // overlap next slab with compute

        const unsigned* a32 = (const unsigned*)lds_a[pb]; // row = 32 uints
        const unsigned* b32 = (const unsigned*)lds_b[pb]; // row = 32 uints
#pragma unroll
        for (int ks = 0; ks < 2; ++ks) {
            Frag af[2], bf[4];
#pragma unroll
            for (int i = 0; i < 2; ++i)
                af[i] = load_afrag(a32, (wm * 32 + i * 16 + lm) * 32 + ks * 16, half);
#pragma unroll
            for (int j = 0; j < 4; ++j)
                bf[j] = load_bfrag(b32, (wn * 64 + j * 16 + lm) * 32 + ks * 16, half);
#pragma unroll
            for (int i = 0; i < 2; ++i)
#pragma unroll
                for (int j = 0; j < 4; ++j)
                    acc[i][j] = __builtin_amdgcn_wmma_f32_16x16x32_bf16(
                        false, af[i].v, false, bf[j].v, (short)0, acc[i][j], false, false);
        }
        wait_async_copies();
        __syncthreads();
    }

    // ---- epilogue: C layout: VGPR r -> row (r + 8*half), lane lm -> col ----
#pragma unroll
    for (int i = 0; i < 2; ++i) {
#pragma unroll
        for (int j = 0; j < 4; ++j) {
            int gm_base = bm0 + wm * 32 + i * 16 + half * 8;
            int gn      = bn0 + wn * 64 + j * 16 + lm;
#pragma unroll
            for (int r = 0; r < 8; ++r) {
                int gm = gm_base + r;
                float v = acc[i][j][r];
                if (residual) v += residual[(size_t)gm * N + gn];
                if (relu)     v = fmaxf(v, 0.0f);
                if (outF)     outF[(size_t)gm * N + gn] = v;
                if (outB) {
                    size_t idx;
                    if (permute_qkv) {
                        int bb = gm >> 11, s = gm & (SEQ - 1);
                        int hh = gn >> 6,  dd = gn & (HD - 1);
                        idx = (((size_t)bb * NH + hh) * SEQ + s) * HD + dd;
                    } else {
                        idx = (size_t)gm * N + gn;
                    }
                    outB[idx] = bf16rne(v);
                }
            }
        }
    }
}

// ---------------- flash attention (bf16 WMMA, online softmax) ----------------
#define QT 128
#define KT 64

__global__ __launch_bounds__(256)
void flash_attn_wmma(const unsigned short* __restrict__ Q,  // [B,H,S,HD] bf16
                     const unsigned short* __restrict__ Kt, // [B,H,S,HD]
                     const unsigned short* __restrict__ V,  // [B,H,S,HD]
                     unsigned short* __restrict__ O,        // [B,S,DIM] bf16 (merged heads)
                     float scale)
{
    __shared__ __attribute__((aligned(16))) unsigned short lds_k[KT * HD];      // [key][d]
    __shared__ __attribute__((aligned(16))) unsigned short lds_v[HD * KT];      // [d][key]
    __shared__ __attribute__((aligned(16))) unsigned short lds_p[8 * 16 * KT];  // per-wave P [m][key]

    const int b = blockIdx.z, h = blockIdx.y;
    const int q0   = blockIdx.x * QT;
    const int tid  = threadIdx.x;
    const int wave = tid >> 5;
    const int lane = tid & 31;
    const int lm   = lane & 15;
    const int half = lane >> 4;

    const size_t headOff = ((size_t)b * NH + h) * SEQ * HD;
    const unsigned short* qg = Q + headOff;
    const unsigned short* kg = Kt + headOff;
    const unsigned short* vg = V + headOff;

    // Q fragments for this wave's 16 rows (K-dim = head dim = 64 -> 2 frags).
    Frag qf[2];
    {
        const unsigned* g = (const unsigned*)(qg + (size_t)(q0 + wave * 16 + lm) * HD);
#pragma unroll
        for (int f = 0; f < 2; ++f)
#pragma unroll
            for (int j = 0; j < 4; ++j) {
                qf[f].u[j]     = g[f * 16 + half * 4 + j];
                qf[f].u[4 + j] = g[f * 16 + 8 + half * 4 + j];
            }
    }

    float m_run[8], l_run[8];
    v8f o_acc[4];
#pragma unroll
    for (int r = 0; r < 8; ++r) { m_run[r] = -1e30f; l_run[r] = 0.0f; }
#pragma unroll
    for (int j = 0; j < 4; ++j) o_acc[j] = vzero8();

    unsigned short* pw = lds_p + wave * (16 * KT);
    const unsigned* p32 = (const unsigned*)pw;          // row = 32 uints
    const unsigned* k32 = (const unsigned*)lds_k;       // row = 32 uints
    const unsigned* v32 = (const unsigned*)lds_v;       // row = 32 uints

    for (int kb = 0; kb < SEQ; kb += KT) {
        // ---- stage K tile [key][d] via async b128 copies (512 slots) ----
        {
            uint4* sk = (uint4*)lds_k;
#pragma unroll
            for (int e = 0; e < 2; ++e) {
                int s   = e * 256 + tid;
                int row = s >> 3, ch = s & 7;
                async_copy_b128(&sk[s], kg + ((size_t)(kb + row) * HD + ch * 8));
            }
        }
        // ---- stage V tile -> [d][key] with interleaved key-pair b32 stores ----
        {
            int kp = tid >> 3;           // 32 key-pairs
            int d0 = (tid & 7) * 8;      // 8 d-groups
            union { uint4 q; unsigned short s[8]; } r0, r1;
            r0.q = *(const uint4*)(vg + (size_t)(kb + 2 * kp)     * HD + d0);
            r1.q = *(const uint4*)(vg + (size_t)(kb + 2 * kp + 1) * HD + d0);
            unsigned* sv = (unsigned*)lds_v;
#pragma unroll
            for (int i = 0; i < 8; ++i)
                sv[(d0 + i) * (KT / 2) + kp] =
                    (unsigned)r0.s[i] | ((unsigned)r1.s[i] << 16);
        }
        wait_async_copies();
        __syncthreads();

        // ---- S = Q @ K^T (4 key-subtiles x 2 K-steps over d) ----
        v8f sacc[4];
#pragma unroll
        for (int j = 0; j < 4; ++j) {
            Frag bk0 = load_bfrag(k32, (j * 16 + lm) * 32 + 0,  half);
            Frag bk1 = load_bfrag(k32, (j * 16 + lm) * 32 + 16, half);
            v8f s = vzero8();
            s = __builtin_amdgcn_wmma_f32_16x16x32_bf16(false, qf[0].v, false, bk0.v, (short)0, s, false, false);
            s = __builtin_amdgcn_wmma_f32_16x16x32_bf16(false, qf[1].v, false, bk1.v, (short)0, s, false, false);
#pragma unroll
            for (int r = 0; r < 8; ++r) s[r] *= scale;
            sacc[j] = s;
        }

        // ---- online softmax (rows r and r+8 live in the two 16-lane halves) ----
#pragma unroll
        for (int r = 0; r < 8; ++r) {
            float mx = sacc[0][r];
#pragma unroll
            for (int j = 1; j < 4; ++j) mx = fmaxf(mx, sacc[j][r]);
            mx = fmaxf(mx, __shfl_xor(mx, 1, 32));
            mx = fmaxf(mx, __shfl_xor(mx, 2, 32));
            mx = fmaxf(mx, __shfl_xor(mx, 4, 32));
            mx = fmaxf(mx, __shfl_xor(mx, 8, 32));
            float mnew  = fmaxf(m_run[r], mx);
            float alpha = __expf(m_run[r] - mnew);
            float rs = 0.0f;
#pragma unroll
            for (int j = 0; j < 4; ++j) {
                float pv = __expf(sacc[j][r] - mnew);
                rs += pv;
                pw[(r + 8 * half) * KT + j * 16 + lm] = bf16rne(pv);
            }
            rs += __shfl_xor(rs, 1, 32);
            rs += __shfl_xor(rs, 2, 32);
            rs += __shfl_xor(rs, 4, 32);
            rs += __shfl_xor(rs, 8, 32);
            l_run[r] = l_run[r] * alpha + rs;
            m_run[r] = mnew;
#pragma unroll
            for (int j = 0; j < 4; ++j) o_acc[j][r] *= alpha;
        }

        // ---- O += P @ V ----
        Frag pf[2];
        pf[0] = load_afrag(p32, lm * 32 + 0,  half);
        pf[1] = load_afrag(p32, lm * 32 + 16, half);
#pragma unroll
        for (int j = 0; j < 4; ++j) {
            Frag bv0 = load_bfrag(v32, (j * 16 + lm) * 32 + 0,  half);
            Frag bv1 = load_bfrag(v32, (j * 16 + lm) * 32 + 16, half);
            o_acc[j] = __builtin_amdgcn_wmma_f32_16x16x32_bf16(false, pf[0].v, false, bv0.v, (short)0, o_acc[j], false, false);
            o_acc[j] = __builtin_amdgcn_wmma_f32_16x16x32_bf16(false, pf[1].v, false, bv1.v, (short)0, o_acc[j], false, false);
        }
        __syncthreads();
    }

    // ---- normalize and store merged-head output [b][s][h*64+d] ----
#pragma unroll
    for (int j = 0; j < 4; ++j) {
#pragma unroll
        for (int r = 0; r < 8; ++r) {
            int m = r + 8 * half;
            int srow = q0 + wave * 16 + m;
            int d = j * 16 + lm;
            float val = o_acc[j][r] / l_run[r];
            O[((size_t)b * SEQ + srow) * DIM + h * HD + d] = bf16rne(val);
        }
    }
}

// ---------------- orchestration ----------------
extern "C" void kernel_launch(void* const* d_in, const int* in_sizes, int n_in,
                              void* d_out, int out_size, void* d_ws, size_t ws_size,
                              hipStream_t stream) {
    const float* x  = (const float*)d_in[0];
    const float* wq = (const float*)d_in[1];
    const float* wk = (const float*)d_in[2];
    const float* wv = (const float*)d_in[3];
    const float* wo = (const float*)d_in[4];
    const float* w1 = (const float*)d_in[5];
    const float* w2 = (const float*)d_in[6];
    float* out = (float*)d_out;

    size_t off = 0;
    auto take = [&](size_t bytes) -> void* {
        void* p = (char*)d_ws + off;
        off += (bytes + 255) & ~(size_t)255;
        return p;
    };
    unsigned short* xb  = (unsigned short*)take((size_t)TOK * DIM * 2);
    unsigned short* wqb = (unsigned short*)take((size_t)DIM * DIM * 2);
    unsigned short* wkb = (unsigned short*)take((size_t)DIM * DIM * 2);
    unsigned short* wvb = (unsigned short*)take((size_t)DIM * DIM * 2);
    unsigned short* wob = (unsigned short*)take((size_t)DIM * DIM * 2);
    unsigned short* w1b = (unsigned short*)take((size_t)DIM * HDIM * 2);
    unsigned short* w2b = (unsigned short*)take((size_t)HDIM * DIM * 2);
    unsigned short* qb  = (unsigned short*)take((size_t)TOK * DIM * 2);
    unsigned short* kb  = (unsigned short*)take((size_t)TOK * DIM * 2);
    unsigned short* vb  = (unsigned short*)take((size_t)TOK * DIM * 2);
    unsigned short* ab  = (unsigned short*)take((size_t)TOK * DIM * 2);
    float*          x1f = (float*)take((size_t)TOK * DIM * 4);
    unsigned short* x1b = (unsigned short*)take((size_t)TOK * DIM * 2);
    unsigned short* hb  = (unsigned short*)take((size_t)TOK * HDIM * 2);

    auto cast8 = [&](const float* src, unsigned short* dst, size_t n) {
        int n8 = (int)(n / 8);
        cast_f32_bf16_v8<<<(n8 + 255) / 256, 256, 0, stream>>>(
            (const float4*)src, (uint4*)dst, n8);
    };

    // 1) casts to bf16
    cast8(x,  xb,  (size_t)TOK * DIM);
    cast8(wq, wqb, (size_t)DIM * DIM);
    cast8(wk, wkb, (size_t)DIM * DIM);
    cast8(wv, wvb, (size_t)DIM * DIM);
    cast8(wo, wob, (size_t)DIM * DIM);
    cast8(w1, w1b, (size_t)DIM * HDIM);
    cast8(w2, w2b, (size_t)HDIM * DIM);

    // 2) QKV projections (store head-major for attention)
    dim3 gP(DIM / BN, TOK / BM);
    gemm_bf16_wmma<<<gP, 256, 0, stream>>>(xb, wqb, nullptr, nullptr, qb, TOK, DIM, DIM, 0, 1);
    gemm_bf16_wmma<<<gP, 256, 0, stream>>>(xb, wkb, nullptr, nullptr, kb, TOK, DIM, DIM, 0, 1);
    gemm_bf16_wmma<<<gP, 256, 0, stream>>>(xb, wvb, nullptr, nullptr, vb, TOK, DIM, DIM, 0, 1);

    // 3) flash attention
    flash_attn_wmma<<<dim3(SEQ / QT, NH, BATCH), 256, 0, stream>>>(qb, kb, vb, ab, 0.125f);

    // 4) x1 = x + attn @ wo   (f32 residual path + bf16 copy for MLP)
    gemm_bf16_wmma<<<gP, 256, 0, stream>>>(ab, wob, x, x1f, x1b, TOK, DIM, DIM, 0, 0);

    // 5) h = relu(x1 @ w1)
    dim3 gUp(HDIM / BN, TOK / BM);
    gemm_bf16_wmma<<<gUp, 256, 0, stream>>>(x1b, w1b, nullptr, nullptr, hb, TOK, HDIM, DIM, 1, 0);

    // 6) out = x1 + h @ w2
    gemm_bf16_wmma<<<gP, 256, 0, stream>>>(hb, w2b, x1f, out, nullptr, TOK, DIM, HDIM, 0, 0);
}
